// CvtLstm_66941360276248
// MI455X (gfx1250) — matrix operand order repaired and checked by reading, not compiled, not run on hardware
//
#include <hip/hip_runtime.h>
#include <math.h>

typedef __bf16 bf16;
typedef __attribute__((ext_vector_type(16))) __bf16 v16bf;
typedef __attribute__((ext_vector_type(8)))  __bf16 v8bf;
typedef __attribute__((ext_vector_type(8)))  float  v8f;

#define DEV __device__ __forceinline__

// ---- problem constants ----
constexpr int NN   = 8;            // batch
constexpr int CI   = 128;          // input channels
constexpr int RR   = 256;          // recurrent channels
constexpr int HWP  = 576;          // 24*24 pixels per image
constexpr int CO   = 256;          // output channels
constexpr int NPIX = NN * HWP;     // 4608 total pixels
constexpr int DVAL = 484;          // 22*22 valid K/V tokens
constexpr int DP   = 512;          // padded K/V tokens
constexpr int KCONV= RR * 9;       // 2304 im2col depth
constexpr int KZ   = 1536;         // gate GEMM depth: 4*A + 2*R
constexpr int RNP  = RR * NPIX;

// =======================================================================
// WMMA fragment helpers (wave32; A/B use this file's contiguous k order, same for both, not the ISA table order)
// A 16x32 bf16: lane m=l&15; elems e -> K = half*16 + e
// B 32x16 bf16: lane n=l&15; elems e -> K = half*16 + e
// C/D 16x16 f32: vgpr r -> row = r + half*8, col = l&15
// =======================================================================
DEV int laneid() { return (int)(threadIdx.x & 31u); }

DEV v16bf a_frag_load(const bf16* A, int lda, int m0, int k0) {
  int l = laneid();
  const bf16* row = A + (size_t)(m0 + (l & 15)) * lda + k0 + ((l >> 4) << 4);
  v8bf lo = *(const v8bf*)(row);
  v8bf hi = *(const v8bf*)(row + 8);
  v16bf a;
#pragma unroll
  for (int i = 0; i < 8; ++i) { a[i] = lo[i]; a[i + 8] = hi[i]; }
  return a;
}

// B stored transposed (N x K row-major): lane reads 16 contiguous K
DEV v16bf b_frag_nmajor(const bf16* BT, int ldt, int k0, int n0) {
  int l = laneid();
  return *(const v16bf*)(BT + (size_t)(n0 + (l & 15)) * ldt + k0 + ((l >> 4) << 4));
}

// B fragment from async-staged LDS tile bt[32 k][16 n] (k-major, scalar gather)
DEV v16bf b_frag_lds(const bf16 (*bt)[16]) {
  int l = laneid();
  int kb = (l >> 4) << 4, n = l & 15;
  v16bf b;
#pragma unroll
  for (int e = 0; e < 16; ++e) b[e] = bt[kb + e][n];
  return b;
}

// B fragment from transposed LDS tile bt[16 n][40 k-padded]: 2x ds_load_b128
DEV v16bf b_frag_lds_t(const bf16 (*bt)[40]) {
  int l = laneid();
  const bf16* p = &bt[l & 15][(l >> 4) << 4];
  v8bf lo = *(const v8bf*)(p);
  v8bf hi = *(const v8bf*)(p + 8);
  v16bf b;
#pragma unroll
  for (int i = 0; i < 8; ++i) { b[i] = lo[i]; b[i + 8] = hi[i]; }
  return b;
}

DEV v8f wmma_bf(v16bf a, v16bf b, v8f c) {
  return __builtin_amdgcn_wmma_f32_16x16x32_bf16(false, a, false, b, (short)0, c,
                                                 false, false);
}

DEV float sigmoidf(float x) { return 1.0f / (1.0f + __expf(-x)); }

// =======================================================================
// Cooperative GEMM strip, async variant: 8 waves share one 32x16 B tile,
// staged global->LDS with CDNA5 async-to-LDS (1 B32 per lane = 1KB tile).
// Used for the small input GEMM (K=128).
// =======================================================================
DEV v8f gemm_strip_async(const bf16* __restrict__ A, int K, const bf16* __restrict__ B,
                         int n0, int m0, bf16 (*bt)[16]) {
  v8f acc = {};
  int tid = threadIdx.x;
  int kk = tid >> 3, np = tid & 7;
  for (int ks = 0; ks < K / 32; ++ks) {
    const bf16* gp = B + (size_t)(ks * 32 + kk) * NPIX + n0 + 2 * np;
    unsigned lp = (unsigned)(size_t)&bt[kk][2 * np];   // low 32 bits = LDS offset
    asm volatile("global_load_async_to_lds_b32 %0, %1, off"
                 :: "v"(lp), "v"((unsigned long long)(size_t)gp) : "memory");
    asm volatile("s_wait_asynccnt 0x0" ::: "memory");
    __syncthreads();
    v16bf af = a_frag_load(A, K, m0, ks * 32);
    v16bf bf = b_frag_lds(bt);
    acc = wmma_bf(af, bf, acc);
    __syncthreads();
  }
  return acc;
}

// =======================================================================
// Cooperative GEMM strip, transposed-tile variant: coalesced global reads,
// transposed ds_store_b16 into bt[16][40]; fragments = 2x ds_load_b128.
// Used for the heavy gate / output GEMMs.
// =======================================================================
DEV v8f gemm_strip_t(const bf16* __restrict__ A, int K, const bf16* __restrict__ B,
                     int n0, int m0, bf16 (*bt)[40]) {
  v8f acc = {};
  int tid = threadIdx.x;
  int kk = tid >> 3, np = tid & 7;
  for (int ks = 0; ks < K / 32; ++ks) {
    const bf16* gp = B + (size_t)(ks * 32 + kk) * NPIX + n0 + 2 * np;
    bf16 v0 = gp[0], v1 = gp[1];
    bt[2 * np][kk]     = v0;
    bt[2 * np + 1][kk] = v1;
    __syncthreads();
    v16bf af = a_frag_load(A, K, m0, ks * 32);
    v16bf bf = b_frag_lds_t(bt);
    acc = wmma_bf(af, bf, acc);
    __syncthreads();
  }
  return acc;
}

// =======================================================================
// Prep / cast kernels
// =======================================================================
__global__ void k_cast_flat(const float* __restrict__ s, bf16* __restrict__ d, int n) {
  for (int i = blockIdx.x * blockDim.x + threadIdx.x; i < n; i += gridDim.x * blockDim.x)
    d[i] = (bf16)s[i];
}

__global__ void k_zero_bf(bf16* __restrict__ d, int n) {
  for (int i = blockIdx.x * blockDim.x + threadIdx.x; i < n; i += gridDim.x * blockDim.x)
    d[i] = (bf16)0.0f;
}

// conv weight [O][C][3][3] (o*2304 + c*9 + t)  ->  [O][t*256 + c]
__global__ void k_permute_convw(const float* __restrict__ s, bf16* __restrict__ d) {
  int idx = blockIdx.x * blockDim.x + threadIdx.x;
  if (idx >= 256 * KCONV) return;
  int o = idx / KCONV, rem = idx - o * KCONV;
  int t = rem >> 8, c = rem & 255;
  d[idx] = (bf16)s[o * KCONV + c * 9 + t];
}

// Wcat[g][o][k]: k<1024 -> W_tok[g][j][o][c]; k>=1024 -> W_skip[g][j][o][c]
__global__ void k_prep_wcat(const float* __restrict__ wtok, const float* __restrict__ wskip,
                            bf16* __restrict__ d) {
  int idx = blockIdx.x * blockDim.x + threadIdx.x;
  if (idx >= 4 * 256 * KZ) return;
  int gm = idx / KZ;                 // g*256 + o
  int k  = idx - gm * KZ;
  int g = gm >> 8, o = gm & 255;
  float v;
  if (k < 1024) {
    int j = k >> 8, c = k & 255;
    v = wtok[(((size_t)(g * 4 + j) * 256) + o) * 256 + c];
  } else {
    int k2 = k - 1024;
    int j = k2 >> 8, c = k2 & 255;
    v = wskip[(((size_t)(g * 2 + j) * 256) + o) * 256 + c];
  }
  d[idx] = (bf16)v;
}

// x0 [n][i][hw] -> [i][n*576+hw]
__global__ void k_cast_x0(const float* __restrict__ s, bf16* __restrict__ d) {
  int idx = blockIdx.x * blockDim.x + threadIdx.x;
  if (idx >= CI * NPIX) return;
  int i = idx / NPIX, p = idx - i * NPIX;
  int n = p / HWP, hw = p - n * HWP;
  d[idx] = (bf16)s[((size_t)n * CI + i) * HWP + hw];
}

// h_prev [n][c][hw] -> zbuf rows 1280.. : [c][n*576+hw]
__global__ void k_cast_h(const float* __restrict__ s, bf16* __restrict__ d) {
  int idx = blockIdx.x * blockDim.x + threadIdx.x;
  if (idx >= RR * NPIX) return;
  int c = idx / NPIX, p = idx - c * NPIX;
  int n = p / HWP, hw = p - n * HWP;
  d[idx] = (bf16)s[((size_t)n * RR + c) * HWP + hw];
}

// =======================================================================
// Input 1x1 conv: x = tanh(W_in @ x0 + b_in)  -> zbuf rows 1024..1279 (bf16)
// =======================================================================
__global__ void k_gemm_in(const bf16* __restrict__ A, const bf16* __restrict__ B,
                          const float* __restrict__ bias, bf16* __restrict__ out) {
  __shared__ alignas(16) bf16 bt[32][16];
  int wid = threadIdx.x >> 5;
  int bm = blockIdx.x & 1, bn = blockIdx.x >> 1;     // 2 x 288
  int m0 = bm * 128 + wid * 16, n0 = bn * 16;
  v8f acc = gemm_strip_async(A, CI, B, n0, m0, bt);
  int l = laneid(); int half = l >> 4; int p = n0 + (l & 15);
#pragma unroll
  for (int r = 0; r < 8; ++r) {
    int m = m0 + r + half * 8;
    out[(size_t)m * NPIX + p] = (bf16)tanhf(acc[r] + bias[m]);
  }
}

// =======================================================================
// 3x3 conv as implicit GEMM.  Block = 8 waves (8 M-tiles) sharing one pixel
// tile; im2col B tile staged transposed in LDS (pad logic per element).
// mode 0: SAME (q) -> [n][g][hw][c]        (pixel-major, A-side of attn)
// mode 1: VALID (v) -> [n][g][c][512]      (channel-major, B-side of AV)
// mode 2: VALID (k) -> [n][g][512][c]      (pixel-major, B-side of QK^T)
// =======================================================================
__global__ void k_conv3(const bf16* __restrict__ Wm,   // [256][2304] (o, t*256+c)
                        const bf16* __restrict__ src,  // [256][NPIX]
                        bf16* __restrict__ outq,
                        bf16* __restrict__ outkv,
                        int mode) {
  __shared__ alignas(16) bf16 bt[16][40];
  int wid = threadIdx.x >> 5;
  int bid = blockIdx.x;
  int mg = bid & 1; bid >>= 1;                 // M half (0..1)
  int PT = mode ? 31 : 36;
  int pt = bid % PT;
  int n  = bid / PT;
  int mt = mg * 8 + wid;
  int l = laneid();
  v8f acc = {};
  for (int ks = 0; ks < KCONV / 32; ++ks) {
    // ---- cooperative im2col stage: 512 elems, 2 per thread, transposed ----
    for (int e = threadIdx.x; e < 512; e += 256) {
      int k = e >> 4, nn2 = e & 15;
      int kglob = ks * 32 + k;
      int t = kglob >> 8, c = kglob & 255;
      int ky = t / 3, kx = t - ky * 3;
      int col = pt * 16 + nn2;
      bf16 v = (bf16)0.0f;
      if (mode == 0) {
        int y = col / 24, x = col - y * 24;
        int sy = y + ky - 1, sx = x + kx - 1;
        if (sy >= 0 && sy < 24 && sx >= 0 && sx < 24)
          v = src[(size_t)c * NPIX + n * HWP + sy * 24 + sx];
      } else if (col < DVAL) {
        int oy = col / 22, ox = col - oy * 22;
        v = src[(size_t)c * NPIX + n * HWP + (oy + ky) * 24 + ox + kx];
      }
      bt[nn2][k] = v;
    }
    __syncthreads();
    v16bf af = a_frag_load(Wm, KCONV, mt * 16, ks * 32);
    v16bf bf = b_frag_lds_t(bt);
    if (ks + 1 < KCONV / 32)
      __builtin_prefetch(Wm + (size_t)(mt * 16 + (l & 15)) * KCONV + (ks + 1) * 32, 0, 1);
    acc = wmma_bf(af, bf, acc);
    __syncthreads();
  }
  int half = l >> 4;
  int col = pt * 16 + (l & 15);
#pragma unroll
  for (int r = 0; r < 8; ++r) {
    int ch = mt * 16 + r + half * 8;
    int g = ch >> 5, c = ch & 31;
    if (mode == 0) {
      outq[((size_t)((n * 8 + g) * HWP + col)) * 32 + c] = (bf16)acc[r];
    } else if (col < DVAL) {
      if (mode == 1)
        outkv[((size_t)((n * 8 + g) * 32 + c)) * DP + col] = (bf16)acc[r];
      else
        outkv[((size_t)((n * 8 + g) * DP + col)) * 32 + c] = (bf16)acc[r];
    }
  }
}

// =======================================================================
// Attention: one wave per (a,b,n,g,qtile).  S = q·kᵀ, softmax, out = P·vᵀ
// q: [a][n][g][576][32] pixel-major; k: [b][n][g][512][32] pixel-major;
// v: [b][n][g][32][512] channel-major.
// =======================================================================
__global__ void k_attn(const bf16* __restrict__ qpm, const bf16* __restrict__ kpm,
                       const bf16* __restrict__ vcm, bf16* __restrict__ zbuf) {
  __shared__ float S[16][DP];
  __shared__ alignas(32) bf16 P[16][DP];
  __shared__ float rinv[16];
  int bid = blockIdx.x;
  int qt = bid % 36; bid /= 36;
  int g  = bid % 8;  bid /= 8;
  int n  = bid % 8;  bid /= 8;
  int b  = bid & 1;
  int a  = bid >> 1;
  int l = laneid(); int half = l >> 4; int coln = l & 15;

  const bf16* qbase = qpm + (size_t)((a * 8 + n) * 8 + g) * HWP * 32;
  const bf16* kbase = kpm + (size_t)((b * 8 + n) * 8 + g) * DP * 32;
  const bf16* vbase = vcm + (size_t)((b * 8 + n) * 8 + g) * 32 * DP;

  // ---- GEMM1: S[q][d] = sum_c q[q][c] * k[c][d]  (single K=32 step) ----
  v16bf qf = a_frag_load(qbase, 32, qt * 16, 0);
  for (int dt = 0; dt < 31; ++dt) {
    v16bf kf = b_frag_nmajor(kbase, 32, 0, dt * 16);   // contiguous 32B per lane
    v8f s = {};
    s = wmma_bf(qf, kf, s);
    int d = dt * 16 + coln;
    if (d < DVAL) {
#pragma unroll
      for (int r = 0; r < 8; ++r) S[r + half * 8][d] = s[r];
    }
  }
  __syncthreads();

  // ---- softmax over d<484, unnormalized exp -> P (bf16), 1/sum -> rinv ----
  for (int row = 0; row < 16; ++row) {
    float m = -3.0e38f;
    for (int d = l; d < DVAL; d += 32) m = fmaxf(m, S[row][d]);
#pragma unroll
    for (int off = 16; off; off >>= 1) m = fmaxf(m, __shfl_xor(m, off, 32));
    float sum = 0.f;
    for (int d = l; d < DP; d += 32) {
      float e = (d < DVAL) ? __expf(S[row][d] - m) : 0.f;
      P[row][d] = (bf16)e;
      sum += e;
    }
#pragma unroll
    for (int off = 16; off; off >>= 1) sum += __shfl_xor(sum, off, 32);
    if (l == 0) rinv[row] = 1.0f / sum;
  }
  __syncthreads();

  // ---- GEMM2: out[q][c] = sum_d P[q][d] * v[c][d], scaled by rinv[q] ----
  int j = a * 2 + b;
  for (int ct = 0; ct < 2; ++ct) {
    v8f acc = {};
#pragma unroll
    for (int ks = 0; ks < DP / 32; ++ks) {
      v16bf af = a_frag_load(&P[0][0], DP, 0, ks * 32);
      v16bf vf = b_frag_nmajor(vbase, DP, ks * 32, ct * 16);
      acc = wmma_bf(af, vf, acc);
    }
    int c = ct * 16 + coln;
    size_t chrow = (size_t)(j * 256 + g * 32 + c) * NPIX;
    int pbase = n * HWP + qt * 16;
#pragma unroll
    for (int r = 0; r < 8; ++r) {
      int row = r + half * 8;
      zbuf[chrow + pbase + row] = (bf16)(acc[r] * rinv[row]);
    }
  }
}

// =======================================================================
// Gate GEMM: pre = Wcat[1024][1536] @ Z[1536][4608] + b_gate  (fp32 out)
// =======================================================================
__global__ void k_gemm_gate(const bf16* __restrict__ A, const bf16* __restrict__ B,
                            const float* __restrict__ bias, float* __restrict__ out) {
  __shared__ alignas(16) bf16 bt[16][40];
  int wid = threadIdx.x >> 5;
  int bm = blockIdx.x & 7, bn = blockIdx.x >> 3;     // 8 x 288
  int m0 = bm * 128 + wid * 16, n0 = bn * 16;
  v8f acc = gemm_strip_t(A, KZ, B, n0, m0, bt);
  int l = laneid(); int half = l >> 4; int p = n0 + (l & 15);
#pragma unroll
  for (int r = 0; r < 8; ++r) {
    int m = m0 + r + half * 8;
    out[(size_t)m * NPIX + p] = acc[r] + bias[m];
  }
}

// =======================================================================
// LSTM elementwise: c = f*c_prev + i*g ; h = o*tanh(c)  -> bf16 [ch][pix]
// =======================================================================
__global__ void k_lstm(const float* __restrict__ pre, const float* __restrict__ cprev,
                       bf16* __restrict__ hn) {
  int idx = blockIdx.x * blockDim.x + threadIdx.x;
  if (idx >= RNP) return;
  int ch = idx / NPIX, p = idx - ch * NPIX;
  int n = p / HWP, hw = p - n * HWP;
  float iv = sigmoidf(pre[idx]);
  float fv = sigmoidf(pre[RNP + idx]);
  float gv = tanhf(pre[2 * RNP + idx]);
  float ov = sigmoidf(pre[3 * RNP + idx]);
  float cp = cprev[((size_t)n * RR + ch) * HWP + hw];
  float c = fv * cp + iv * gv;
  hn[idx] = (bf16)(ov * tanhf(c));
}

// =======================================================================
// Output 1x1 conv: out = W_out @ h_new + b_out  -> fp32 NCHW d_out
// =======================================================================
__global__ void k_gemm_out(const bf16* __restrict__ A, const bf16* __restrict__ B,
                           const float* __restrict__ bias, float* __restrict__ out) {
  __shared__ alignas(16) bf16 bt[16][40];
  int wid = threadIdx.x >> 5;
  int bm = blockIdx.x & 1, bn = blockIdx.x >> 1;     // 2 x 288
  int m0 = bm * 128 + wid * 16, n0 = bn * 16;
  v8f acc = gemm_strip_t(A, RR, B, n0, m0, bt);
  int l = laneid(); int half = l >> 4; int p = n0 + (l & 15);
  int n = p / HWP, hw = p - n * HWP;
#pragma unroll
  for (int r = 0; r < 8; ++r) {
    int m = m0 + r + half * 8;
    out[((size_t)n * CO + m) * HWP + hw] = acc[r] + bias[m];
  }
}

// =======================================================================
// Host launcher
// =======================================================================
extern "C" void kernel_launch(void* const* d_in, const int* in_sizes, int n_in,
                              void* d_out, int out_size, void* d_ws, size_t ws_size,
                              hipStream_t stream) {
  (void)in_sizes; (void)n_in; (void)out_size; (void)ws_size;
  const float* x0     = (const float*)d_in[0];
  const float* hprev  = (const float*)d_in[1];
  const float* cprev  = (const float*)d_in[2];
  const float* W_in   = (const float*)d_in[3];
  const float* b_in   = (const float*)d_in[4];
  const float* Wcv6[6]= {(const float*)d_in[5], (const float*)d_in[6],
                         (const float*)d_in[7], (const float*)d_in[8],
                         (const float*)d_in[9], (const float*)d_in[10]};
  const float* W_tok  = (const float*)d_in[11];
  const float* b_gate = (const float*)d_in[12];
  const float* W_skip = (const float*)d_in[13];
  const float* W_out  = (const float*)d_in[14];
  const float* b_out  = (const float*)d_in[15];
  float* out = (float*)d_out;

  char* wsp = (char*)d_ws;
  auto alloc = [&](size_t bytes) -> char* {
    char* p = wsp; wsp += (bytes + 255) & ~(size_t)255; return p;
  };
  bf16* x0bf = (bf16*)alloc((size_t)CI * NPIX * 2);
  bf16* zbuf = (bf16*)alloc((size_t)KZ * NPIX * 2);       // [1536][4608]
  bf16* qbf  = (bf16*)alloc((size_t)2 * 64 * HWP * 32 * 2);
  bf16* kbf  = (bf16*)alloc((size_t)2 * 64 * DP * 32 * 2);   // pixel-major
  bf16* vbf  = (bf16*)alloc((size_t)2 * 64 * 32 * DP * 2);   // channel-major
  bf16* wcv  = (bf16*)alloc((size_t)6 * 256 * KCONV * 2);
  bf16* winb = (bf16*)alloc((size_t)RR * CI * 2);
  bf16* wcat = (bf16*)alloc((size_t)4 * 256 * KZ * 2);
  bf16* woutb= (bf16*)alloc((size_t)CO * RR * 2);
  float* pre = (float*)alloc((size_t)4 * RNP * 4);
  bf16* hn   = (bf16*)alloc((size_t)RNP * 2);

  bf16* xz = zbuf + (size_t)1024 * NPIX;   // x rows of Z
  bf16* hz = zbuf + (size_t)1280 * NPIX;   // h rows of Z

  // ---- weight & input prep ----
  k_cast_flat<<<(RR * CI + 255) / 256, 256, 0, stream>>>(W_in, winb, RR * CI);
  k_cast_flat<<<(CO * RR + 255) / 256, 256, 0, stream>>>(W_out, woutb, CO * RR);
  for (int w = 0; w < 6; ++w)
    k_permute_convw<<<(256 * KCONV + 255) / 256, 256, 0, stream>>>(
        Wcv6[w], wcv + (size_t)w * 256 * KCONV);
  k_prep_wcat<<<(4 * 256 * KZ + 255) / 256, 256, 0, stream>>>(W_tok, W_skip, wcat);
  k_cast_x0<<<(CI * NPIX + 255) / 256, 256, 0, stream>>>(x0, x0bf);
  k_cast_h<<<(RR * NPIX + 255) / 256, 256, 0, stream>>>(hprev, hz);
  k_zero_bf<<<4096, 256, 0, stream>>>(kbf, 2 * 2 * 64 * 32 * DP);  // kbf+vbf adjacent

  // ---- stage 1: input conv (tanh) ----
  k_gemm_in<<<2 * 288, 256, 0, stream>>>(winb, x0bf, b_in, xz);

  // ---- stage 2: six 3x3 convs (block = 2 Mgroups x pixel tile x image) ----
  const size_t WSZ = (size_t)256 * KCONV;
  const size_t QBR = (size_t)64 * HWP * 32;   // per-branch q size
  const size_t KVB = (size_t)64 * 32 * DP;    // per-branch k/v size
  int gq = 2 * 36 * 8, gkv = 2 * 31 * 8;
  k_conv3<<<gq, 256, 0, stream>>>(wcv + 0 * WSZ, xz, qbf,        nullptr,   0); // q_x
  k_conv3<<<gq, 256, 0, stream>>>(wcv + 1 * WSZ, hz, qbf + QBR,  nullptr,   0); // q_h
  k_conv3<<<gkv, 256, 0, stream>>>(wcv + 2 * WSZ, xz, nullptr, kbf,         2); // k_x
  k_conv3<<<gkv, 256, 0, stream>>>(wcv + 3 * WSZ, hz, nullptr, kbf + KVB,   2); // k_h
  k_conv3<<<gkv, 256, 0, stream>>>(wcv + 4 * WSZ, xz, nullptr, vbf,         1); // v_x
  k_conv3<<<gkv, 256, 0, stream>>>(wcv + 5 * WSZ, hz, nullptr, vbf + KVB,   1); // v_h

  // ---- stage 3: attention (writes z_tok rows 0..1023 of Z) ----
  k_attn<<<2 * 2 * 8 * 8 * 36, 32, 0, stream>>>(qbf, kbf, vbf, zbuf);

  // ---- stage 4: gates, LSTM cell, output conv ----
  k_gemm_gate<<<8 * 288, 256, 0, stream>>>(wcat, zbuf, b_gate, pre);
  k_lstm<<<(RNP + 255) / 256, 256, 0, stream>>>(pre, cprev, hn);
  k_gemm_out<<<2 * 288, 256, 0, stream>>>(woutb, hn, b_out, out);
}
